// STGCN_11132555231484
// MI455X (gfx1250) — matrix operand order repaired and checked by reading, not compiled, not run on hardware
//
#include <hip/hip_runtime.h>
#include <hip/hip_bf16.h>

typedef __attribute__((ext_vector_type(16))) __bf16 v16bf;
typedef __attribute__((ext_vector_type(8)))  float  v8f;

#define THREADS 256

// ---------------- gcn_norm precompute ----------------

__global__ void k_fill(float* __restrict__ p, float v, int n) {
  int i = blockIdx.x * blockDim.x + threadIdx.x;
  if (i < n) p[i] = v;
}

__global__ void k_deg_acc(const int* __restrict__ dst, const float* __restrict__ w,
                          float* __restrict__ deg, int E) {
  int e = blockIdx.x * blockDim.x + threadIdx.x;
  if (e < E) atomicAdd(&deg[dst[e]], w[e]);
}

__global__ void k_dinv(float* __restrict__ dn, int n) {
  int i = blockIdx.x * blockDim.x + threadIdx.x;
  if (i < n) {
    float d = dn[i];
    dn[i] = d > 0.f ? rsqrtf(d) : 0.f;
  }
}

__global__ void k_norm(const int* __restrict__ src, const int* __restrict__ dst,
                       const float* __restrict__ w, const float* __restrict__ dinv,
                       float* __restrict__ nrm, int E) {
  int e = blockIdx.x * blockDim.x + threadIdx.x;
  if (e < E) nrm[e] = dinv[src[e]] * w[e] * dinv[dst[e]];
}

// Convert W (row-major K x F, fp32) -> transposed bf16 panel Wt[c*K + k].
__global__ void k_w2bf(const float* __restrict__ W, __bf16* __restrict__ Wt,
                       int K, int F) {
  int idx = blockIdx.x * blockDim.x + threadIdx.x;
  if (idx < K * F) {
    int k = idx / F, c = idx % F;
    Wt[c * K + k] = (__bf16)W[idx];
  }
}

// ---------------- WMMA GEMM: T[n x F] = A[n x 128] @ W[128 x F] ----------------
// bf16 inputs, f32 accumulate via v_wmma_f32_16x16x32_bf16.
// The pre-transposed bf16 weight panel (Wg, [F][K]) is staged to LDS via the
// Tensor Data Mover (tensor_load_to_lds + s_wait_tensorcnt) when available,
// so each lane's B fragment is one contiguous 32-byte LDS read.
// A fragments use contiguous k order (not the ISA table order), same as B:
// lane L (row M = L&15, g = L>>4): elem e -> K = kb + g*16 + e.
// B fragment (lane L, col c): elem e -> K = kb + g*16 + e.

__device__ inline void cvt8(v16bf& a, int base, float4 lo, float4 hi) {
  a[base + 0] = (__bf16)lo.x; a[base + 1] = (__bf16)lo.y;
  a[base + 2] = (__bf16)lo.z; a[base + 3] = (__bf16)lo.w;
  a[base + 4] = (__bf16)hi.x; a[base + 5] = (__bf16)hi.y;
  a[base + 6] = (__bf16)hi.z; a[base + 7] = (__bf16)hi.w;
}

template <int F>
__global__ __launch_bounds__(256) void k_gemm(const float* __restrict__ A,
                                              const __bf16* __restrict__ Wg,
                                              float* __restrict__ T, int n) {
  constexpr int K = 128;
  __shared__ __attribute__((aligned(32))) __bf16 Wt[F * K];

  int tid = threadIdx.x;

#if __has_builtin(__builtin_amdgcn_tensor_load_to_lds)
  // ---- TDM staging: one DMA of the whole F*K bf16 panel (1D, 8B elements) ----
  if (tid < 32) {   // one wave issues the descriptor; EXEC is ignored by TDM
    typedef __attribute__((ext_vector_type(4))) unsigned int u32x4;
    typedef __attribute__((ext_vector_type(4))) int i32x4;
    typedef __attribute__((ext_vector_type(8))) int i32x8;
    constexpr unsigned NE8 = (unsigned)(F * K * 2 / 8);  // # of 8-byte elements
    unsigned long long ga = (unsigned long long)Wg;
    u32x4 g0 = { 1u,                               // count=1 (valid D#)
                 0u,                               // lds_addr = 0 (Wt base)
                 (unsigned)ga,                     // global_addr[31:0]
                 (unsigned)((ga >> 32) & 0x01FFFFFFull) | 0x80000000u }; // type=2
    i32x8 g1 = { (int)(3u << 16),                  // data_size=3 (8 bytes)
                 (int)((NE8 & 0xFFFFu) << 16),     // tensor_dim0[15:0]
                 (int)((NE8 >> 16) | (1u << 16)),  // tensor_dim0[31:16], tensor_dim1=1
                 (int)(NE8 << 16),                 // tile_dim0 = NE8
                 0,                                // tile_dim1/2 unused
                 (int)NE8,                         // tensor_dim0_stride
                 0, 0 };
    i32x4 gz = { 0, 0, 0, 0 };
#if __clang_major__ >= 23
    i32x8 gz8 = { 0, 0, 0, 0, 0, 0, 0, 0 };
    __builtin_amdgcn_tensor_load_to_lds(g0, g1, gz, gz, gz8, 0);
#else
    __builtin_amdgcn_tensor_load_to_lds(g0, g1, gz, gz, 0);
#endif
    __builtin_amdgcn_s_wait_tensorcnt(0);
  }
  __syncthreads();
#else
  // ---- fallback: cooperative 16B-chunk copy ----
  {
    const uint4* s4 = (const uint4*)Wg;
    uint4* d4 = (uint4*)Wt;
    for (int i = tid; i < F * K * 2 / 16; i += 256) d4[i] = s4[i];
  }
  __syncthreads();
#endif

  int wave = tid >> 5;      // wave32
  int lane = tid & 31;
  int g    = lane >> 4;
  int m0   = blockIdx.x * 128 + wave * 16;
  int arow = m0 + (lane & 15);
  if (arow > n - 1) arow = n - 1;      // clamp: keep EXEC all-ones for WMMA
  const float* Arow = A + (long)arow * K;
  bool full = (m0 + 16 <= n);          // wave-uniform fast path

  // Hoist all A fragments (K = 128 -> 4 chunks of 32).
  v16bf afrag[4];
#pragma unroll
  for (int kc = 0; kc < 4; ++kc) {
    int kb = kc * 32;
    float4 a0 = *(const float4*)(Arow + kb + g * 16);
    float4 a1 = *(const float4*)(Arow + kb + g * 16 + 4);
    float4 a2 = *(const float4*)(Arow + kb + g * 16 + 8);
    float4 a3 = *(const float4*)(Arow + kb + g * 16 + 12);
    cvt8(afrag[kc], 0, a0, a1);
    cvt8(afrag[kc], 8, a2, a3);
  }

#pragma unroll
  for (int j = 0; j < F / 32; ++j) {
    int c0 = j * 32 + (lane & 15);
    int c1 = c0 + 16;
    // Load all 8 B fragments for this pass into distinct registers first,
    // then issue the 8 WMMAs back-to-back (acc0/acc1 alternation hides the
    // WMMA->WMMA RAW hazard; one dscnt wait instead of one per WMMA).
    v16bf bf[8];
#pragma unroll
    for (int kc = 0; kc < 4; ++kc) {
      int kb = kc * 32;
      bf[2 * kc + 0] = *(const v16bf*)&Wt[c0 * K + kb + g * 16];
      bf[2 * kc + 1] = *(const v16bf*)&Wt[c1 * K + kb + g * 16];
    }
    v8f acc0 = {};
    v8f acc1 = {};
#pragma unroll
    for (int kc = 0; kc < 4; ++kc) {
      acc0 = __builtin_amdgcn_wmma_f32_16x16x32_bf16(
          false, afrag[kc], false, bf[2 * kc + 0], (short)0, acc0, false, false);
      acc1 = __builtin_amdgcn_wmma_f32_16x16x32_bf16(
          false, afrag[kc], false, bf[2 * kc + 1], (short)0, acc1, false, false);
    }
    // C/D layout: VGPR r holds row M = g*8 + r, col N = lane&15.
    if (full) {
#pragma unroll
      for (int r = 0; r < 8; ++r) {
        long row = m0 + g * 8 + r;
        T[row * F + c0] = acc0[r];
        T[row * F + c1] = acc1[r];
      }
    } else {
#pragma unroll
      for (int r = 0; r < 8; ++r) {
        int row = m0 + g * 8 + r;
        if (row < n) {
          T[(long)row * F + c0] = acc0[r];
          T[(long)row * F + c1] = acc1[r];
        }
      }
    }
  }
}

// ---------------- aggregation ----------------

// G[i][f] = b[f] + dinv[i]^2 * T[i][f]   (bias + self-loop term)
__global__ void k_agg_init(const float* __restrict__ T, const float* __restrict__ dinv,
                           const float* __restrict__ b, float* __restrict__ G,
                           int n, int F) {
  int fq = F >> 2;
  int i4 = blockIdx.x * blockDim.x + threadIdx.x;
  if (i4 >= n * fq) return;
  int node = i4 / fq;
  int f4 = (i4 - node * fq) * 4;
  float dv = dinv[node];
  float sn = dv * dv;
  float4 t  = *(const float4*)(T + (long)node * F + f4);
  float4 bb = *(const float4*)(b + f4);
  float4 o;
  o.x = bb.x + sn * t.x; o.y = bb.y + sn * t.y;
  o.z = bb.z + sn * t.z; o.w = bb.w + sn * t.w;
  *(float4*)(G + (long)node * F + f4) = o;
}

// F/4 lanes per edge: G[dst] += norm * T[src]; float4 per lane, f32 atomics.
template <int F>
__global__ void k_scatter(const float* __restrict__ T, const int* __restrict__ src,
                          const int* __restrict__ dst, const float* __restrict__ nrm,
                          float* __restrict__ G, int E) {
  constexpr int LPE = F / 4;               // lanes per edge (32 or 16)
  int gtid = blockIdx.x * blockDim.x + threadIdx.x;
  int e    = gtid / LPE;
  int l    = gtid % LPE;
  if (e >= E) return;
  int s = src[e], d = dst[e];
  float nv = nrm[e];
  float4 v = *(const float4*)(T + (long)s * F + l * 4);
  float* out = G + (long)d * F + l * 4;
  atomicAdd(out + 0, v.x * nv);
  atomicAdd(out + 1, v.y * nv);
  atomicAdd(out + 2, v.z * nv);
  atomicAdd(out + 3, v.w * nv);
}

__global__ void k_relu4(const float4* __restrict__ in, float4* __restrict__ out, int n4) {
  int i = blockIdx.x * blockDim.x + threadIdx.x;
  if (i < n4) {
    float4 v = in[i];
    float4 o;
    o.x = fmaxf(v.x, 0.f); o.y = fmaxf(v.y, 0.f);
    o.z = fmaxf(v.z, 0.f); o.w = fmaxf(v.w, 0.f);
    out[i] = o;
  }
}

__global__ void k_sigmoid4(const float4* __restrict__ in, float4* __restrict__ out, int n4) {
  int i = blockIdx.x * blockDim.x + threadIdx.x;
  if (i < n4) {
    float4 v = in[i];
    float4 o;
    o.x = 1.f / (1.f + expf(-v.x)); o.y = 1.f / (1.f + expf(-v.y));
    o.z = 1.f / (1.f + expf(-v.z)); o.w = 1.f / (1.f + expf(-v.w));
    out[i] = o;
  }
}

// ---------------- launcher ----------------

static inline int cdiv(int a, int b) { return (a + b - 1) / b; }

extern "C" void kernel_launch(void* const* d_in, const int* in_sizes, int n_in,
                              void* d_out, int out_size, void* d_ws, size_t ws_size,
                              hipStream_t stream) {
  const int IN = 128, H = 128, OUT = 64;
  const int N = in_sizes[0] / IN;
  const int E = in_sizes[2];

  const float* x   = (const float*)d_in[0];
  const int*   ei  = (const int*)d_in[1];
  const int*   src = ei;          // edge_index[0]
  const int*   dst = ei + E;      // edge_index[1]
  const float* ew  = (const float*)d_in[2];
  const float* W1  = (const float*)d_in[3];  const float* b1 = (const float*)d_in[4];
  const float* W2  = (const float*)d_in[5];  const float* b2 = (const float*)d_in[6];
  const float* W3  = (const float*)d_in[7];  const float* b3 = (const float*)d_in[8];
  const float* W4  = (const float*)d_in[9];  const float* b4 = (const float*)d_in[10];
  float* out = (float*)d_out;

  // Workspace layout (floats), 256B-aligned chunks.
  auto align64 = [](long v) { return (v + 63) & ~63L; };
  float* ws  = (float*)d_ws;
  long off = 0;
  float* dn  = ws + off; off += align64(N);        // degree -> dinv (in place)
  float* nrm = ws + off; off += align64(E);        // per-edge norm
  float* T   = ws + off; off += (long)N * H;       // pre-aggregation h@W
  float* G   = ws + off; off += (long)N * H;       // aggregated
  float* Hh  = ws + off; off += (long)N * H;       // activated
  __bf16* W1t = (__bf16*)(ws + off); off += (H * H) / 2;    // bf16 panels
  __bf16* W2t = (__bf16*)(ws + off); off += (H * H) / 2;
  __bf16* W3t = (__bf16*)(ws + off); off += (H * H) / 2;
  __bf16* W4t = (__bf16*)(ws + off); off += (H * OUT) / 2;
  (void)ws_size;

  // ---- gcn_norm (shared by all 4 layers) + weight panel conversion ----
  k_fill<<<cdiv(N, THREADS), THREADS, 0, stream>>>(dn, 1.0f, N);   // self-loop weight
  k_deg_acc<<<cdiv(E, THREADS), THREADS, 0, stream>>>(dst, ew, dn, E);
  k_dinv<<<cdiv(N, THREADS), THREADS, 0, stream>>>(dn, N);
  k_norm<<<cdiv(E, THREADS), THREADS, 0, stream>>>(src, dst, ew, dn, nrm, E);
  k_w2bf<<<cdiv(H * H, THREADS), THREADS, 0, stream>>>(W1, W1t, H, H);
  k_w2bf<<<cdiv(H * H, THREADS), THREADS, 0, stream>>>(W2, W2t, H, H);
  k_w2bf<<<cdiv(H * H, THREADS), THREADS, 0, stream>>>(W3, W3t, H, H);
  k_w2bf<<<cdiv(H * OUT, THREADS), THREADS, 0, stream>>>(W4, W4t, H, OUT);

  const int gemmBlocks = cdiv(N, 128);
  const int elemsH     = N * H;
  const int elemsO     = N * OUT;

  // ---- layer 1 ----
  k_gemm<128><<<gemmBlocks, 256, 0, stream>>>(x, W1t, T, N);
  k_agg_init<<<cdiv(N * (H / 4), THREADS), THREADS, 0, stream>>>(T, dn, b1, G, N, H);
  k_scatter<128><<<cdiv(E * 32, THREADS), THREADS, 0, stream>>>(T, src, dst, nrm, G, E);
  k_relu4<<<cdiv(elemsH / 4, THREADS), THREADS, 0, stream>>>((const float4*)G, (float4*)Hh, elemsH / 4);

  // ---- layer 2 ----
  k_gemm<128><<<gemmBlocks, 256, 0, stream>>>(Hh, W2t, T, N);
  k_agg_init<<<cdiv(N * (H / 4), THREADS), THREADS, 0, stream>>>(T, dn, b2, G, N, H);
  k_scatter<128><<<cdiv(E * 32, THREADS), THREADS, 0, stream>>>(T, src, dst, nrm, G, E);
  k_relu4<<<cdiv(elemsH / 4, THREADS), THREADS, 0, stream>>>((const float4*)G, (float4*)Hh, elemsH / 4);

  // ---- layer 3 ----
  k_gemm<128><<<gemmBlocks, 256, 0, stream>>>(Hh, W3t, T, N);
  k_agg_init<<<cdiv(N * (H / 4), THREADS), THREADS, 0, stream>>>(T, dn, b3, G, N, H);
  k_scatter<128><<<cdiv(E * 32, THREADS), THREADS, 0, stream>>>(T, src, dst, nrm, G, E);
  k_relu4<<<cdiv(elemsH / 4, THREADS), THREADS, 0, stream>>>((const float4*)G, (float4*)Hh, elemsH / 4);

  // ---- layer 4 (OUT=64, sigmoid -> d_out) ----
  k_gemm<64><<<gemmBlocks, 256, 0, stream>>>(Hh, W4t, T, N);
  k_agg_init<<<cdiv(N * (OUT / 4), THREADS), THREADS, 0, stream>>>(T, dn, b4, G, N, OUT);
  k_scatter<64><<<cdiv(E * 16, THREADS), THREADS, 0, stream>>>(T, src, dst, nrm, G, E);
  k_sigmoid4<<<cdiv(elemsO / 4, THREADS), THREADS, 0, stream>>>((const float4*)G, (float4*)out, elemsO / 4);
}
